// ODE_RNN_3255585210718
// MI455X (gfx1250) — compile-verified, matrix operand order checked by reading, not run on hardware
//
#include <hip/hip_runtime.h>
#include <hip/hip_bf16.h>
#include <math.h>

// ---------------------------------------------------------------------------
// Fused ODE-RNN step for MI455X (gfx1250, wave32, WMMA bf16 -> f32).
// Each 256-thread WG (8 waves) owns a 16-row batch tile and performs the full
// RK4 + GRU + readout pipeline without spilling activations to HBM.
// Weights are pre-packed (one-time per call) into WMMA-B fragment layout so a
// B operand is one 32-byte contiguous global load per lane.
// ---------------------------------------------------------------------------

typedef __attribute__((ext_vector_type(16))) __bf16 v16bf;
typedef __attribute__((ext_vector_type(8)))  __bf16 v8bf;
typedef __attribute__((ext_vector_type(8)))  float  v8f;

#define K_OUT 16
#define K_H   1024
#define K_H2  2048
#define K_IN  256
#define ROWW  (K_OUT + K_H)   // 1040 floats per state row
#define DT    0.1f

#define M_TILE    16
#define SA_STRIDE 1040        // bf16 elements (pad for LDS banking)
#define ST_STRIDE 136
#define SX_STRIDE 272

__device__ __forceinline__ __bf16 f2bf(float f) {
  union { float f; unsigned u; } x; x.f = f;
  unsigned r = x.u + 0x7FFFu + ((x.u >> 16) & 1u);   // round-to-nearest-even
  unsigned short h = (unsigned short)(r >> 16);
  return __builtin_bit_cast(__bf16, h);
}

// A-fragment (16xK, bf16, 16x16x32 step) from row-major LDS tile.
// lane&15 = M row; lane>>4 selects K-halves {kh..kh+7} U {16+kh..16+kh+7}.
__device__ __forceinline__ v16bf load_a(const __bf16* s, int stride, int kb, int lane) {
  int m  = lane & 15;
  int kh = (lane >> 4) << 3;
  const __bf16* p = s + m * stride + kb + kh;
  v8bf lo = *(const v8bf*)p;          // K = kb+kh .. +7   (ds_load_b128)
  v8bf hi = *(const v8bf*)(p + 16);   // K = kb+16+kh .. +7
  return __builtin_shufflevector(lo, hi, 0,1,2,3,4,5,6,7,8,9,10,11,12,13,14,15);
}

// B-fragment from packed weights: 512 bf16 per fragment, lane-contiguous.
__device__ __forceinline__ v16bf load_b(const __bf16* frag, int lane) {
  return *(const v16bf*)(frag + lane * 16);
}

#define WMMA_BF16(a,b,c) \
  __builtin_amdgcn_wmma_f32_16x16x32_bf16(false,(a),false,(b),(short)0,(c),false,false)

// acc[0..7]: 16x16 tiles over an 8-tile (128 col) strip starting at ntBase.
__device__ __forceinline__ void gemm_acc8(v8f acc[8], const __bf16* A, int astride,
                                          const __bf16* Bp, int KT, int ntBase, int lane) {
  for (int kt = 0; kt < KT; ++kt) {
    v16bf a = load_a(A, astride, kt * 32, lane);
#pragma unroll
    for (int i = 0; i < 8; ++i) {
      v16bf b = load_b(Bp + ((size_t)(ntBase + i) * KT + kt) * 512, lane);
      acc[i] = WMMA_BF16(a, b, acc[i]);
    }
  }
}

// ---------------------------------------------------------------------------
// One-time weight pack: f32 (K,N) [or (N,K) if transposed] -> bf16 fragments.
// Fragment (nt,kt) is 512 contiguous bf16 at ((nt*KT)+kt)*512; within it,
// lane l holds 16 values for column n = nt*16+(l&15), K set matching load_a.
// ---------------------------------------------------------------------------
__global__ void pack_b_kernel(const float* __restrict__ src, __bf16* __restrict__ dst,
                              int K, int N, int ld, int transposed) {
  long long total = (long long)K * N;
  long long t = (long long)blockIdx.x * blockDim.x + threadIdx.x;
  if (t >= total) return;
  int frag = (int)(t >> 9);
  int rr   = (int)(t & 511);
  int lane = rr >> 4, j = rr & 15;
  int KT = K >> 5;
  int kt = frag % KT, nt = frag / KT;
  int n  = nt * 16 + (lane & 15);
  int kh = (lane >> 4) << 3;
  int k  = kt * 32 + ((j < 8) ? (kh + j) : (16 + kh + (j - 8)));
  float v = transposed ? src[(size_t)n * ld + k] : src[(size_t)k * ld + n];
  dst[t] = f2bf(v);
}

// ---------------------------------------------------------------------------
// Fused main kernel.
// ---------------------------------------------------------------------------
__global__ __launch_bounds__(256)
void ode_rnn_fused_kernel(const float* __restrict__ state,
                          const float* __restrict__ xt,
                          const float* __restrict__ b1,
                          const float* __restrict__ b2,
                          const float* __restrict__ bih,
                          const float* __restrict__ bhh,
                          const float* __restrict__ bly,
                          const int*   __restrict__ y_type,
                          const __bf16* __restrict__ P1,   // W1   packed (K=1024,N=2048)
                          const __bf16* __restrict__ P2,   // W2   packed (K=2048,N=1024)
                          const __bf16* __restrict__ Pih,  // Wih^T packed (K=256, N=3072)
                          const __bf16* __restrict__ Phh,  // Whh^T packed (K=1024,N=3072)
                          const __bf16* __restrict__ Ply,  // Wly  packed (K=1024,N=16)
                          float* __restrict__ out) {
  __shared__ __bf16 sA[M_TILE * SA_STRIDE];  // current h (bf16 A tile)
  __shared__ __bf16 sT[M_TILE * ST_STRIDE];  // tanh chunk (16x128)
  __shared__ __bf16 sX[M_TILE * SX_STRIDE];  // x tile (16x256)

  const int tid  = threadIdx.x;
  const int wave = tid >> 5;
  const int lane = tid & 31;
  const int lanelow = lane & 15;
  const int hi8  = (lane >> 4) << 3;         // C-frag: row m = r + hi8
  const int row0 = blockIdx.x * M_TILE;

  // ---- init: h0 strip to registers, h0 -> sA (bf16), x -> sX (bf16) ----
  v8f h0r[8];
#pragma unroll
  for (int f = 0; f < 8; ++f) {
    int col = wave * 128 + f * 16 + lanelow;
#pragma unroll
    for (int r = 0; r < 8; ++r) {
      float v = state[(size_t)(row0 + r + hi8) * ROWW + K_OUT + col];
      h0r[f][r] = v;
      sA[(r + hi8) * SA_STRIDE + col] = f2bf(v);
    }
  }
  for (int idx = tid; idx < M_TILE * K_IN; idx += 256) {
    int m = idx >> 8, k = idx & 255;
    sX[m * SX_STRIDE + k] = f2bf(xt[(size_t)(row0 + m) * K_IN + k]);
  }
  __syncthreads();

  // ---- RK4: 4 evaluations of k = tanh(h@W1+b1)@W2+b2, fused in LDS ----
  v8f ksum[8];
#pragma unroll
  for (int f = 0; f < 8; ++f) ksum[f] = v8f{};

  for (int s = 0; s < 4; ++s) {
    v8f acc2[8];
#pragma unroll
    for (int f = 0; f < 8; ++f) acc2[f] = v8f{};

    for (int kc = 0; kc < 16; ++kc) {           // 128-col chunks of the 2048 mid dim
      // GEMM1: this wave computes 16x16 tile (cols kc*128 + wave*16) over K=1024
      v8f acc1 = v8f{};
      int nt1 = kc * 8 + wave;
      for (int kt = 0; kt < 32; ++kt) {
        v16bf a = load_a(sA, SA_STRIDE, kt * 32, lane);
        v16bf b = load_b(P1 + ((size_t)nt1 * 32 + kt) * 512, lane);
        acc1 = WMMA_BF16(a, b, acc1);
      }
      __syncthreads();                          // sT(prev chunk) fully consumed
      {
        int gcol = kc * 128 + wave * 16 + lanelow;
        float bb = b1[gcol];
#pragma unroll
        for (int r = 0; r < 8; ++r)
          sT[(r + hi8) * ST_STRIDE + wave * 16 + lanelow] = f2bf(tanhf(acc1[r] + bb));
      }
      __syncthreads();
      // GEMM2 accumulate: t(16x128) @ W2[chunk, wave-strip(128)]
#pragma unroll
      for (int kt2 = 0; kt2 < 4; ++kt2) {
        v16bf a = load_a(sT, ST_STRIDE, kt2 * 32, lane);
        int ktg = kc * 4 + kt2;
#pragma unroll
        for (int nt = 0; nt < 8; ++nt) {
          v16bf b = load_b(P2 + ((size_t)(wave * 8 + nt) * 64 + ktg) * 512, lane);
          acc2[nt] = WMMA_BF16(a, b, acc2[nt]);
        }
      }
    }
    __syncthreads();                            // all GEMM1 reads of sA done
    float wsum = (s == 1 || s == 2) ? 2.f : 1.f;
    float cn   = (s < 2) ? (0.5f * DT) : DT;
#pragma unroll
    for (int f = 0; f < 8; ++f) {
      int col = wave * 128 + f * 16 + lanelow;
      float b2v = b2[col];
#pragma unroll
      for (int r = 0; r < 8; ++r) {
        float kv = acc2[f][r] + b2v;
        ksum[f][r] += wsum * kv;
        if (s < 3)
          sA[(r + hi8) * SA_STRIDE + col] = f2bf(h0r[f][r] + cn * kv);  // next stage h
      }
    }
    __syncthreads();
  }

  // ---- h' = h0 + dt/6 * sum ; keep f32 in ksum, bf16 tile in sA ----
#pragma unroll
  for (int f = 0; f < 8; ++f) {
    int col = wave * 128 + f * 16 + lanelow;
#pragma unroll
    for (int r = 0; r < 8; ++r) {
      float hp = h0r[f][r] + (DT / 6.0f) * ksum[f][r];
      ksum[f][r] = hp;
      sA[(r + hi8) * SA_STRIDE + col] = f2bf(hp);
    }
  }
  __syncthreads();

  // ---- GRU cell on the wave's 128-col strip ----
  const int cb = wave * 128;
  const int nb = cb >> 4;

  v8f rg[8], hn[8], xn[8];
#pragma unroll
  for (int f = 0; f < 8; ++f) { rg[f] = v8f{}; hn[f] = v8f{}; xn[f] = v8f{}; }
  gemm_acc8(rg, sX, SX_STRIDE, Pih, 8,  (0    >> 4) + nb, lane);   // i_r
  gemm_acc8(rg, sA, SA_STRIDE, Phh, 32, (0    >> 4) + nb, lane);   // + h_r
  gemm_acc8(hn, sA, SA_STRIDE, Phh, 32, (2048 >> 4) + nb, lane);   // h_n
  gemm_acc8(xn, sX, SX_STRIDE, Pih, 8,  (2048 >> 4) + nb, lane);   // i_n
#pragma unroll
  for (int i = 0; i < 8; ++i) {
    int col = cb + i * 16 + lanelow;
    float br   = bih[col] + bhh[col];
    float bn_i = bih[2048 + col];
    float bn_h = bhh[2048 + col];
#pragma unroll
    for (int r = 0; r < 8; ++r) {
      float rv = 1.f / (1.f + expf(-(rg[i][r] + br)));
      xn[i][r] = tanhf(xn[i][r] + bn_i + rv * (hn[i][r] + bn_h));   // n gate
    }
  }
  v8f zg[8];
#pragma unroll
  for (int f = 0; f < 8; ++f) zg[f] = v8f{};
  gemm_acc8(zg, sX, SX_STRIDE, Pih, 8,  (1024 >> 4) + nb, lane);
  gemm_acc8(zg, sA, SA_STRIDE, Phh, 32, (1024 >> 4) + nb, lane);
#pragma unroll
  for (int i = 0; i < 8; ++i) {
    int col = cb + i * 16 + lanelow;
    float bz = bih[1024 + col] + bhh[1024 + col];
#pragma unroll
    for (int r = 0; r < 8; ++r) {
      float zv = 1.f / (1.f + expf(-(zg[i][r] + bz)));
      float hnew = (1.f - zv) * xn[i][r] + zv * ksum[i][r];
      ksum[i][r] = hnew;
      out[(size_t)(row0 + r + hi8) * ROWW + K_OUT + col] = hnew;
    }
  }
  __syncthreads();                       // all h' reads of sA complete
#pragma unroll
  for (int i = 0; i < 8; ++i)
#pragma unroll
    for (int r = 0; r < 8; ++r)
      sA[(r + hi8) * SA_STRIDE + cb + i * 16 + lanelow] = f2bf(ksum[i][r]);
  __syncthreads();

  // ---- readout: ly = h_new @ Wly + bly ; masked by y_type (wave 0 only) ----
  if (wave == 0) {
    v8f acc = v8f{};
    for (int kt = 0; kt < 32; ++kt) {
      v16bf a = load_a(sA, SA_STRIDE, kt * 32, lane);
      v16bf b = load_b(Ply + (size_t)kt * 512, lane);
      acc = WMMA_BF16(a, b, acc);
    }
    int colj = lanelow;
    int use_ly = (y_type[colj] == 0);
    float bl = bly[colj];
#pragma unroll
    for (int r = 0; r < 8; ++r) {
      size_t row = (size_t)(row0 + r + hi8);
      float lyv = acc[r] + bl;
      float yold = state[row * ROWW + colj];
      out[row * ROWW + colj] = use_ly ? lyv : yold;
    }
  }
}

// ---------------------------------------------------------------------------
extern "C" void kernel_launch(void* const* d_in, const int* in_sizes, int n_in,
                              void* d_out, int out_size, void* d_ws, size_t ws_size,
                              hipStream_t stream) {
  const float* state = (const float*)d_in[0];
  const float* xt    = (const float*)d_in[1];
  const float* W1    = (const float*)d_in[2];
  const float* b1    = (const float*)d_in[3];
  const float* W2    = (const float*)d_in[4];
  const float* b2    = (const float*)d_in[5];
  const float* Wih   = (const float*)d_in[6];
  const float* Whh   = (const float*)d_in[7];
  const float* bih   = (const float*)d_in[8];
  const float* bhh   = (const float*)d_in[9];
  const float* Wly   = (const float*)d_in[10];
  const float* bly   = (const float*)d_in[11];
  const int*   ytyp  = (const int*)d_in[12];
  float* out = (float*)d_out;

  // workspace layout: packed bf16 weights (~16.3 MB total)
  __bf16* ws = (__bf16*)d_ws;
  size_t o = 0;
  __bf16* P1  = ws + o; o += (size_t)K_H  * K_H2;   // 1024x2048
  __bf16* P2  = ws + o; o += (size_t)K_H2 * K_H;    // 2048x1024
  __bf16* Pih = ws + o; o += (size_t)K_IN * 3072;   // 256x3072 (Wih^T)
  __bf16* Phh = ws + o; o += (size_t)K_H  * 3072;   // 1024x3072 (Whh^T)
  __bf16* Ply = ws + o; o += (size_t)K_H  * K_OUT;  // 1024x16

  auto blocks = [](long long total) { return (unsigned)((total + 255) / 256); };
  pack_b_kernel<<<blocks((long long)K_H * K_H2), 256, 0, stream>>>(W1,  P1,  K_H,  K_H2, K_H2, 0);
  pack_b_kernel<<<blocks((long long)K_H2 * K_H), 256, 0, stream>>>(W2,  P2,  K_H2, K_H,  K_H,  0);
  pack_b_kernel<<<blocks((long long)K_IN * 3072),256, 0, stream>>>(Wih, Pih, K_IN, 3072, K_IN, 1);
  pack_b_kernel<<<blocks((long long)K_H  * 3072),256, 0, stream>>>(Whh, Phh, K_H,  3072, K_H,  1);
  pack_b_kernel<<<blocks((long long)K_H * K_OUT),256, 0, stream>>>(Wly, Ply, K_H,  K_OUT,K_OUT,0);

  int B = in_sizes[0] / ROWW;            // 32768
  dim3 grid(B / M_TILE), block(256);
  ode_rnn_fused_kernel<<<grid, block, 0, stream>>>(
      state, xt, b1, b2, bih, bhh, bly, ytyp, P1, P2, Pih, Phh, Ply, out);
}
